// MultiHeadAttention_75402445848972
// MI455X (gfx1250) — matrix operand order repaired and checked by reading, not compiled, not run on hardware
//
#include <hip/hip_runtime.h>

// MI455X / gfx1250, wave32. Full causal MHA forward:
//   QKV proj (bf16 WMMA) -> flash attention (bf16 WMMA, online softmax) -> out proj (bf16 WMMA) + bias
// Workspace: Q,K,V in [B][H][S][64] bf16 + ctx in [B][S][1024] bf16 = 64 MiB total.

typedef __attribute__((ext_vector_type(16))) __bf16        v16bf;
typedef __attribute__((ext_vector_type(8)))  float         v8f;
typedef __attribute__((ext_vector_type(4)))  float         v4f;
typedef __attribute__((ext_vector_type(4)))  unsigned int  v4u;

union Frag {
  v16bf v;
  v4u   q[2];
};

__device__ __forceinline__ v8f wmma_bf16(v16bf a, v16bf b, v8f c) {
  // (neg_a, A, neg_b, B, c_mod, C, reuse_a, reuse_b)
  return __builtin_amdgcn_wmma_f32_16x16x32_bf16(false, a, false, b, (short)0, c, false, false);
}

// ---- byte-permute helpers (v_perm_b32) ----
#if __has_builtin(__builtin_amdgcn_perm)
__device__ __forceinline__ unsigned int pack_hi16(unsigned int a, unsigned int b) {
  // result = (b >> 16) | (a & 0xffff0000)
  return __builtin_amdgcn_perm(a, b, 0x07060302u);
}
__device__ __forceinline__ unsigned int pack_lo16(unsigned int a, unsigned int b) {
  // result = (b & 0xffff) | (a << 16)
  return __builtin_amdgcn_perm(a, b, 0x05040100u);
}
#else
__device__ __forceinline__ unsigned int pack_hi16(unsigned int a, unsigned int b) {
  return (b >> 16) | (a & 0xFFFF0000u);
}
__device__ __forceinline__ unsigned int pack_lo16(unsigned int a, unsigned int b) {
  return (b & 0xFFFFu) | (a << 16);
}
#endif

// fp32 -> bf16 (round half up) pair pack: one add per input + one v_perm
__device__ __forceinline__ unsigned int bfp(float lo, float hi) {
  return pack_hi16(__float_as_uint(hi) + 0x8000u, __float_as_uint(lo) + 0x8000u);
}
__device__ __forceinline__ unsigned short bf16s(float f) {
  return (unsigned short)((__float_as_uint(f) + 0x8000u) >> 16);
}

// ---- async global -> LDS copy (gfx1250 GLOBAL_LOAD_ASYNC_TO_LDS_B128) ----
// Probe-learned signature: (v4i addrspace(1)*, v4i addrspace(3)*, imm offset, imm cpol)
#if __has_builtin(__builtin_amdgcn_global_load_async_to_lds_b128) && \
    __has_builtin(__builtin_amdgcn_s_wait_asynccnt)
#define HAVE_ASYNC_LDS 1
typedef int v4i_gcc __attribute__((vector_size(16)));
typedef __attribute__((address_space(1))) v4i_gcc* g_v4i_p;
typedef __attribute__((address_space(3))) v4i_gcc* l_v4i_p;
__device__ __forceinline__ void async_copy16(const void* gptr, void* lptr) {
  __builtin_amdgcn_global_load_async_to_lds_b128((g_v4i_p)gptr, (l_v4i_p)lptr, 0, 0);
}
__device__ __forceinline__ void async_wait0() {
  __builtin_amdgcn_s_wait_asynccnt(0);
}
#else
#define HAVE_ASYNC_LDS 0
__device__ __forceinline__ void async_copy16(const void* gptr, void* lptr) {
  *(v4u*)lptr = *(const v4u*)gptr;
}
__device__ __forceinline__ void async_wait0() {}
#endif

// ---------------------------------------------------------------------------
// Kernel 1: QKV projection.  C[8192,1024] = x[8192,1024] @ W[1024,1024]
// Block tile 64(M) x 128(N), 256 threads = 8 waves (2x4), wave tile 32x32.
// Output scattered to [B][H][S][64] bf16. blockIdx.z selects {Wq,Wk,Wv}.
// ---------------------------------------------------------------------------
__global__ __launch_bounds__(256) void qkv_gemm_kernel(
    const float* __restrict__ x,
    const float* __restrict__ Wq, const float* __restrict__ Wk, const float* __restrict__ Wv,
    unsigned short* __restrict__ Qo, unsigned short* __restrict__ Ko, unsigned short* __restrict__ Vo)
{
  const float* W;
  unsigned short* O;
  if (blockIdx.z == 0)      { W = Wq; O = Qo; }
  else if (blockIdx.z == 1) { W = Wk; O = Ko; }
  else                      { W = Wv; O = Vo; }

  const int tid  = threadIdx.x;
  const int lane = tid & 31;
  const int wave = tid >> 5;
  const int col  = lane & 15;
  const int hi   = lane >> 4;          // 0: lanes 0-15, 1: lanes 16-31
  const int wm   = wave >> 2;          // 0..1
  const int wn   = wave & 3;           // 0..3
  const int rowBase = blockIdx.x * 64;
  const int nBase   = blockIdx.y * 128;

  __shared__ unsigned short ldsA[64][32];    // A tile, row-major [m][k], bf16
  __shared__ unsigned short ldsW[128][32];   // W tile transposed [n][k], bf16

  const v8f vzero = {};
  v8f acc[2][2];
  acc[0][0] = vzero; acc[0][1] = vzero; acc[1][0] = vzero; acc[1][1] = vzero;

  const int ar  = tid >> 2, acg = (tid & 3) * 8;   // A staging: 64 rows x 4 col-groups of 8
  const int wk2 = (tid >> 4) * 2;                  // W staging: even k row
  const int wnb = (tid & 15) * 8;                  // W staging: 8 n per thread

  for (int k0 = 0; k0 < 1024; k0 += 32) {
    __syncthreads();
    {   // stage A (fp32 -> bf16), one ds_store_b128 per thread
      const float* src = x + (size_t)(rowBase + ar) * 1024 + k0 + acg;
      v4f f0 = *(const v4f*)src;
      v4f f1 = *(const v4f*)(src + 4);
      v4u p;
      p.x = bfp(f0.x, f0.y); p.y = bfp(f0.z, f0.w);
      p.z = bfp(f1.x, f1.y); p.w = bfp(f1.z, f1.w);
      *(v4u*)&ldsA[ar][acg] = p;
    }
    {   // stage W transposed (fp32 -> bf16): two adjacent k rows packed per dword
      const float* s0 = W + (size_t)(k0 + wk2) * 1024 + nBase + wnb;
      v4f a0 = *(const v4f*)s0;          v4f a1 = *(const v4f*)(s0 + 4);
      v4f b0 = *(const v4f*)(s0 + 1024); v4f b1 = *(const v4f*)(s0 + 1028);
      if (k0 + 32 < 1024) __builtin_prefetch(s0 + 32 * 1024, 0, 1);
      *(unsigned int*)&ldsW[wnb + 0][wk2] = bfp(a0.x, b0.x);
      *(unsigned int*)&ldsW[wnb + 1][wk2] = bfp(a0.y, b0.y);
      *(unsigned int*)&ldsW[wnb + 2][wk2] = bfp(a0.z, b0.z);
      *(unsigned int*)&ldsW[wnb + 3][wk2] = bfp(a0.w, b0.w);
      *(unsigned int*)&ldsW[wnb + 4][wk2] = bfp(a1.x, b1.x);
      *(unsigned int*)&ldsW[wnb + 5][wk2] = bfp(a1.y, b1.y);
      *(unsigned int*)&ldsW[wnb + 6][wk2] = bfp(a1.z, b1.z);
      *(unsigned int*)&ldsW[wnb + 7][wk2] = bfp(a1.w, b1.w);
    }
    __syncthreads();

    // Fragment loads per CDNA5 16-bit layouts.
    Frag aF[2], bF[2];
    const int koff = hi ? 8 : 0;     // A: lanes16-31 hold K 8..15 / 24..31
#pragma unroll
    for (int mi = 0; mi < 2; ++mi) {
      int r = wm * 32 + mi * 16 + col;
      aF[mi].q[0] = *(const v4u*)&ldsA[r][koff];
      aF[mi].q[1] = *(const v4u*)&ldsA[r][koff + 16];
    }
    const int kkb = hi ? 8 : 0;      // B: lanes16-31 hold K 8..15 / 24..31
#pragma unroll
    for (int ni = 0; ni < 2; ++ni) {
      int n = wn * 32 + ni * 16 + col;
      bF[ni].q[0] = *(const v4u*)&ldsW[n][kkb];
      bF[ni].q[1] = *(const v4u*)&ldsW[n][kkb + 16];
    }
#pragma unroll
    for (int mi = 0; mi < 2; ++mi)
#pragma unroll
      for (int ni = 0; ni < 2; ++ni)
        acc[mi][ni] = wmma_bf16(aF[mi].v, bF[ni].v, acc[mi][ni]);
  }

  // write bf16 to [B=4][H=16][S=2048][64]
#pragma unroll
  for (int mi = 0; mi < 2; ++mi)
#pragma unroll
    for (int ni = 0; ni < 2; ++ni)
#pragma unroll
      for (int vr = 0; vr < 8; ++vr) {
        int row = rowBase + wm * 32 + mi * 16 + vr + hi * 8;   // C: M = vgpr + 8*hi
        int n   = nBase + wn * 32 + ni * 16 + col;             // C: N = lane&15
        int bb = row >> 11, ss = row & 2047;
        int hh = n >> 6,    dd = n & 63;
        O[(((size_t)(bb * 16 + hh) * 2048) + ss) * 64 + dd] = bf16s(acc[mi][ni][vr]);
      }
}

// ---------------------------------------------------------------------------
// Kernel 2: causal flash attention per (b, h, 64-row q tile), 4 waves.
// S^T tiles (rows = keys, cols = q): softmax reductions are in-lane + one
// xor-16 shuffle; P re-layouts through LDS for P@V.  K/V tiles are
// double-buffered; K uses async global->LDS (ASYNCcnt) prefetch of tile kt+1
// issued before computing tile kt.
// ---------------------------------------------------------------------------
__global__ __launch_bounds__(128) void attn_kernel(
    const unsigned short* __restrict__ Q,
    const unsigned short* __restrict__ K,
    const unsigned short* __restrict__ V,
    unsigned short* __restrict__ Ctx)
{
  const int qt   = blockIdx.x;       // 0..31
  const int h    = blockIdx.y;       // 0..15
  const int b    = blockIdx.z;       // 0..3
  const int tid  = threadIdx.x;
  const int lane = tid & 31;
  const int wave = tid >> 5;         // 0..3, wave owns q rows [qbase+16w, +16)
  const int col  = lane & 15;
  const int hi   = lane >> 4;

  __shared__ unsigned short ldsK[2][64][64];    // K tile row-major [k][d], double buffered
  __shared__ unsigned short ldsVT[2][64][64];   // V tile transposed [d][k], double buffered
  __shared__ unsigned short ldsP[4][16][64];    // per-wave P [q][k] bf16
  __shared__ float          red[4][16];         // per-wave per-q broadcast

  const size_t headBase = ((size_t)(b * 16 + h)) * 2048 * 64;
  const int qbase = qt * 64;
  const int qrow  = qbase + wave * 16 + col;    // this lane's q column

  // stage tile kt into buffer buf: K via async copy, V transposed via perm packs
  auto stage_tile = [&](int kt_, int buf_) {
    const unsigned short* kp = K + headBase + (size_t)kt_ * 4096;
    const unsigned short* vp = V + headBase + (size_t)kt_ * 4096;
#pragma unroll
    for (int i = 0; i < 4; ++i) {
      int idx = tid + i * 128;          // 0..511 b128 units
      int r = idx >> 3, c = (idx & 7) * 8;
      async_copy16(kp + r * 64 + c, &ldsK[buf_][r][c]);
    }
#pragma unroll
    for (int i = 0; i < 2; ++i) {
      int idx = tid + i * 128;          // 0..255: two k rows x 8 d per unit
      int r2 = (idx >> 3) * 2, c = (idx & 7) * 8;
      v4u r0 = *(const v4u*)(vp + r2 * 64 + c);
      v4u r1 = *(const v4u*)(vp + (r2 + 1) * 64 + c);
      *(unsigned int*)&ldsVT[buf_][c + 0][r2] = pack_lo16(r1.x, r0.x);
      *(unsigned int*)&ldsVT[buf_][c + 1][r2] = pack_hi16(r1.x, r0.x);
      *(unsigned int*)&ldsVT[buf_][c + 2][r2] = pack_lo16(r1.y, r0.y);
      *(unsigned int*)&ldsVT[buf_][c + 3][r2] = pack_hi16(r1.y, r0.y);
      *(unsigned int*)&ldsVT[buf_][c + 4][r2] = pack_lo16(r1.z, r0.z);
      *(unsigned int*)&ldsVT[buf_][c + 5][r2] = pack_hi16(r1.z, r0.z);
      *(unsigned int*)&ldsVT[buf_][c + 6][r2] = pack_lo16(r1.w, r0.w);
      *(unsigned int*)&ldsVT[buf_][c + 7][r2] = pack_hi16(r1.w, r0.w);
    }
  };

  // Q fragments (loop-invariant): B-operand of S^T = Q^T, d = 8*hi+s / 16+8*hi+(s-8) per lane, as for K.
  Frag bq0, bq1;
  {
    const unsigned short* qp = Q + headBase + (size_t)qrow * 64;
    const int dd = hi ? 8 : 0;
    bq0.q[0] = *(const v4u*)(qp + dd);
    bq0.q[1] = *(const v4u*)(qp + dd + 16);
    bq1.q[0] = *(const v4u*)(qp + 32 + dd);
    bq1.q[1] = *(const v4u*)(qp + 32 + dd + 16);
  }

  const v8f vzero = {};
  v8f accD[4];
#pragma unroll
  for (int i = 0; i < 4; ++i) accD[i] = vzero;
  float m_run = -1.0e30f, l_run = 0.0f;

  stage_tile(0, 0);
  async_wait0();
  __syncthreads();

  for (int kt = 0; kt <= qt; ++kt) {
    const int buf = kt & 1;
    if (kt < qt) stage_tile(kt + 1, buf ^ 1);   // overlap copy of next tile

    // S^T[key][q] = K @ Q^T over HD=64 (two chained WMMAs per 16-key subtile)
    v8f st[4];
    const int koff = hi ? 8 : 0;
#pragma unroll
    for (int n = 0; n < 4; ++n) {
      Frag a0, a1;
      int kr = n * 16 + col;
      a0.q[0] = *(const v4u*)&ldsK[buf][kr][koff];
      a0.q[1] = *(const v4u*)&ldsK[buf][kr][koff + 16];
      a1.q[0] = *(const v4u*)&ldsK[buf][kr][32 + koff];
      a1.q[1] = *(const v4u*)&ldsK[buf][kr][32 + koff + 16];
      v8f c = wmma_bf16(a0.v, bq0.v, vzero);
      st[n]  = wmma_bf16(a1.v, bq1.v, c);
    }

    // scale 1/sqrt(64), causal mask, online softmax over this 64-key tile
    float mloc = -1.0e30f;
#pragma unroll
    for (int n = 0; n < 4; ++n)
#pragma unroll
      for (int vr = 0; vr < 8; ++vr) {
        int key = kt * 64 + n * 16 + vr + hi * 8;
        float s = st[n][vr] * 0.125f;
        if (key > qrow) s = -1.0e30f;
        st[n][vr] = s;
        mloc = fmaxf(mloc, s);
      }
    mloc = fmaxf(mloc, __shfl_xor(mloc, 16));
    float mnew = fmaxf(m_run, mloc);
    float corr = __expf(m_run - mnew);
    float lloc = 0.0f;
#pragma unroll
    for (int n = 0; n < 4; ++n)
#pragma unroll
      for (int vr = 0; vr < 8; ++vr) {
        float p = __expf(st[n][vr] - mnew);
        st[n][vr] = p;
        lloc += p;
      }
    lloc += __shfl_xor(lloc, 16);
    l_run = l_run * corr + lloc;
    m_run = mnew;

    // broadcast per-q rescale factor to all lanes (ctx rows need it)
    if (!hi) red[wave][col] = corr;
    __syncthreads();
#pragma unroll
    for (int dn = 0; dn < 4; ++dn)
#pragma unroll
      for (int vr = 0; vr < 8; ++vr)
        accD[dn][vr] *= red[wave][vr + hi * 8];

    // write P row-major [q][k] in bf16 (8 contiguous probs per lane -> b128)
#pragma unroll
    for (int n = 0; n < 4; ++n) {
      v4u p;
      p.x = bfp(st[n][0], st[n][1]);
      p.y = bfp(st[n][2], st[n][3]);
      p.z = bfp(st[n][4], st[n][5]);
      p.w = bfp(st[n][6], st[n][7]);
      *(v4u*)&ldsP[wave][col][n * 16 + hi * 8] = p;
    }
    __syncthreads();

    // ctx[16q][64d] += P[16q][64k] @ V[64k][64d]
    Frag ap0, ap1;
    ap0.q[0] = *(const v4u*)&ldsP[wave][col][koff];
    ap0.q[1] = *(const v4u*)&ldsP[wave][col][koff + 16];
    ap1.q[0] = *(const v4u*)&ldsP[wave][col][32 + koff];
    ap1.q[1] = *(const v4u*)&ldsP[wave][col][32 + koff + 16];
    const int kkb = hi ? 8 : 0;
#pragma unroll
    for (int dn = 0; dn < 4; ++dn) {
      Frag bv0, bv1;
      int dr = dn * 16 + col;
      bv0.q[0] = *(const v4u*)&ldsVT[buf][dr][kkb];
      bv0.q[1] = *(const v4u*)&ldsVT[buf][dr][kkb + 16];
      bv1.q[0] = *(const v4u*)&ldsVT[buf][dr][32 + kkb];
      bv1.q[1] = *(const v4u*)&ldsVT[buf][dr][32 + kkb + 16];
      accD[dn] = wmma_bf16(ap0.v, bv0.v, accD[dn]);
      accD[dn] = wmma_bf16(ap1.v, bv1.v, accD[dn]);
    }

    // next tile's async K copies must land; all waves done with this buffer
    async_wait0();
    __syncthreads();
  }

  // finalize: divide by running sum, write ctx bf16 to [B][S][1024]
  if (!hi) red[wave][col] = 1.0f / l_run;
  __syncthreads();
#pragma unroll
  for (int dn = 0; dn < 4; ++dn)
#pragma unroll
    for (int vr = 0; vr < 8; ++vr) {
      float v = accD[dn][vr] * red[wave][vr + hi * 8];
      int srow = qbase + wave * 16 + vr + hi * 8;
      int c    = h * 64 + dn * 16 + col;
      Ctx[((size_t)b * 2048 + srow) * 1024 + c] = bf16s(v);
    }
}

// ---------------------------------------------------------------------------
// Kernel 3: output projection. out[8192,1024] = ctx_bf16 @ Wo + bo (fp32 out)
// A tile is already bf16 -> async global->LDS copy.
// ---------------------------------------------------------------------------
__global__ __launch_bounds__(256) void out_proj_kernel(
    const unsigned short* __restrict__ ctx,
    const float* __restrict__ Wo, const float* __restrict__ bo,
    float* __restrict__ out)
{
  const int tid  = threadIdx.x;
  const int lane = tid & 31;
  const int wave = tid >> 5;
  const int col  = lane & 15;
  const int hi   = lane >> 4;
  const int wm   = wave >> 2, wn = wave & 3;
  const int rowBase = blockIdx.x * 64;
  const int nBase   = blockIdx.y * 128;

  __shared__ unsigned short ldsA[64][32];
  __shared__ unsigned short ldsW[128][32];

  const v8f vzero = {};
  v8f acc[2][2];
  acc[0][0] = vzero; acc[0][1] = vzero; acc[1][0] = vzero; acc[1][1] = vzero;

  const int ar  = tid >> 2, acg = (tid & 3) * 8;
  const int wk2 = (tid >> 4) * 2;
  const int wnb = (tid & 15) * 8;

  for (int k0 = 0; k0 < 1024; k0 += 32) {
    __syncthreads();
    {   // A is already bf16 in workspace: async b128 copy
      async_copy16(ctx + (size_t)(rowBase + ar) * 1024 + k0 + acg, &ldsA[ar][acg]);
    }
    {   // stage Wo transposed, paired k rows
      const float* s0 = Wo + (size_t)(k0 + wk2) * 1024 + nBase + wnb;
      v4f a0 = *(const v4f*)s0;          v4f a1 = *(const v4f*)(s0 + 4);
      v4f b0 = *(const v4f*)(s0 + 1024); v4f b1 = *(const v4f*)(s0 + 1028);
      if (k0 + 32 < 1024) __builtin_prefetch(s0 + 32 * 1024, 0, 1);
      *(unsigned int*)&ldsW[wnb + 0][wk2] = bfp(a0.x, b0.x);
      *(unsigned int*)&ldsW[wnb + 1][wk2] = bfp(a0.y, b0.y);
      *(unsigned int*)&ldsW[wnb + 2][wk2] = bfp(a0.z, b0.z);
      *(unsigned int*)&ldsW[wnb + 3][wk2] = bfp(a0.w, b0.w);
      *(unsigned int*)&ldsW[wnb + 4][wk2] = bfp(a1.x, b1.x);
      *(unsigned int*)&ldsW[wnb + 5][wk2] = bfp(a1.y, b1.y);
      *(unsigned int*)&ldsW[wnb + 6][wk2] = bfp(a1.z, b1.z);
      *(unsigned int*)&ldsW[wnb + 7][wk2] = bfp(a1.w, b1.w);
    }
    async_wait0();
    __syncthreads();

    Frag aF[2], bF[2];
    const int koff = hi ? 8 : 0;
#pragma unroll
    for (int mi = 0; mi < 2; ++mi) {
      int r = wm * 32 + mi * 16 + col;
      aF[mi].q[0] = *(const v4u*)&ldsA[r][koff];
      aF[mi].q[1] = *(const v4u*)&ldsA[r][koff + 16];
    }
    const int kkb = hi ? 8 : 0;
#pragma unroll
    for (int ni = 0; ni < 2; ++ni) {
      int n = wn * 32 + ni * 16 + col;
      bF[ni].q[0] = *(const v4u*)&ldsW[n][kkb];
      bF[ni].q[1] = *(const v4u*)&ldsW[n][kkb + 16];
    }
#pragma unroll
    for (int mi = 0; mi < 2; ++mi)
#pragma unroll
      for (int ni = 0; ni < 2; ++ni)
        acc[mi][ni] = wmma_bf16(aF[mi].v, bF[ni].v, acc[mi][ni]);
  }

#pragma unroll
  for (int mi = 0; mi < 2; ++mi)
#pragma unroll
    for (int ni = 0; ni < 2; ++ni)
#pragma unroll
      for (int vr = 0; vr < 8; ++vr) {
        int row = rowBase + wm * 32 + mi * 16 + vr + hi * 8;
        int n   = nBase + wn * 32 + ni * 16 + col;
        out[(size_t)row * 1024 + n] = acc[mi][ni][vr] + bo[n];
      }
}

// ---------------------------------------------------------------------------
extern "C" void kernel_launch(void* const* d_in, const int* in_sizes, int n_in,
                              void* d_out, int out_size, void* d_ws, size_t ws_size,
                              hipStream_t stream) {
  (void)in_sizes; (void)n_in; (void)out_size; (void)ws_size;
  const float* x  = (const float*)d_in[0];
  const float* Wq = (const float*)d_in[1];
  const float* Wk = (const float*)d_in[2];
  const float* Wv = (const float*)d_in[3];
  const float* Wo = (const float*)d_in[4];
  const float* bo = (const float*)d_in[5];
  float* out = (float*)d_out;

  // workspace: 4 bf16 buffers of B*S*D = 8M elements each (64 MiB total)
  const size_t NE = (size_t)4 * 2048 * 1024;
  unsigned short* Qw = (unsigned short*)d_ws;
  unsigned short* Kw = Qw + NE;
  unsigned short* Vw = Kw + NE;
  unsigned short* Cw = Vw + NE;

  qkv_gemm_kernel<<<dim3(128, 8, 3), 256, 0, stream>>>(x, Wq, Wk, Wv, Qw, Kw, Vw);
  attn_kernel<<<dim3(32, 16, 4), 128, 0, stream>>>(Qw, Kw, Vw, Cw);
  out_proj_kernel<<<dim3(128, 8, 1), 256, 0, stream>>>(Cw, Wo, bo, out);
}
